// GNNEncoder_11416023073362
// MI455X (gfx1250) — matrix operand order repaired and checked by reading, not compiled, not run on hardware
//
#include <hip/hip_runtime.h>
#include <hip/hip_bf16.h>
#include <cstdint>

#define N_NODES 50000
#define N_EDGES 1600000
#define HIDDEN  128
#define NUM_GRAPHS 64

typedef _Float16 half16 __attribute__((ext_vector_type(16)));
typedef float    floatx8 __attribute__((ext_vector_type(8)));

__device__ __forceinline__ uint16_t f2h(float f) {
    union { _Float16 h; uint16_t u; } t; t.h = (_Float16)f; return t.u;
}
__device__ __forceinline__ uint32_t pk2h(float a, float b) {
    return (uint32_t)f2h(a) | ((uint32_t)f2h(b) << 16);
}
__device__ __forceinline__ float gelu(float x) {
    return 0.5f * x * (1.0f + erff(x * 0.70710678118654752f));
}
__device__ __forceinline__ floatx8 wmma16(const uint32_t au[8], const uint32_t bu[8], floatx8 c) {
    union U { half16 v; uint32_t u[8]; };
    U A, B;
#pragma unroll
    for (int i = 0; i < 8; i++) { A.u[i] = au[i]; B.u[i] = bu[i]; }
    return __builtin_amdgcn_wmma_f32_16x16x32_f16(false, A.v, false, B.v, (short)0, c, false, false);
}
// load one B fragment (8 dwords per lane, contiguous, 32B aligned) as 2x b128
__device__ __forceinline__ void ldB(const uint32_t* base, uint32_t bu[8]) {
    const uint4* p = (const uint4*)base;
    uint4 x = p[0], y = p[1];
    bu[0] = x.x; bu[1] = x.y; bu[2] = x.z; bu[3] = x.w;
    bu[4] = y.x; bu[5] = y.y; bu[6] = y.z; bu[7] = y.w;
}
// load one A fragment 32-chunk from a row-major f16 row (per ISA A layout the two
// 4-VGPR halves are 16B-contiguous runs at +hi*8 and +16+hi*8 elements)
__device__ __forceinline__ void ldA(const uint16_t* row, int kc, int hi, uint32_t a[8]) {
    uint4 x = *(const uint4*)(row + kc * 32 + hi * 8);
    uint4 y = *(const uint4*)(row + kc * 32 + 16 + hi * 8);
    a[0] = x.x; a[1] = x.y; a[2] = x.z; a[3] = x.w;
    a[4] = y.x; a[5] = y.y; a[6] = y.z; a[7] = y.w;
}

__global__ void k_zero(float* p, int n) {
    int i = blockIdx.x * blockDim.x + threadIdx.x;
    if (i < n) p[i] = 0.f;
}

__global__ void k_count(const int* __restrict__ batch, float* __restrict__ cnt) {
    int i = blockIdx.x * blockDim.x + threadIdx.x;
    if (i < N_NODES) atomicAdd(&cnt[batch[i]], 1.0f);
}

// h = GELU(LN(x @ W + b)); one wave per node, 4 features per lane
__global__ void k_embed(const float* __restrict__ x, const float* __restrict__ W,
                        const float* __restrict__ b, const float* __restrict__ g,
                        const float* __restrict__ be, float* __restrict__ h,
                        uint16_t* __restrict__ hb) {
    int wid = (blockIdx.x * blockDim.x + threadIdx.x) >> 5;
    int lane = threadIdx.x & 31;
    if (wid >= N_NODES) return;
    const float* xr = x + (size_t)wid * 30;
    float y[4];
#pragma unroll
    for (int f = 0; f < 4; f++) {
        int n = lane + f * 32;
        float acc = b[n];
        for (int k = 0; k < 30; k++) acc += xr[k] * W[k * HIDDEN + n];
        y[f] = acc;
    }
    float s = y[0] + y[1] + y[2] + y[3];
    float s2 = y[0]*y[0] + y[1]*y[1] + y[2]*y[2] + y[3]*y[3];
#pragma unroll
    for (int m = 1; m < 32; m <<= 1) { s += __shfl_xor(s, m); s2 += __shfl_xor(s2, m); }
    float mean = s * (1.f / 128.f);
    float var  = s2 * (1.f / 128.f) - mean * mean;
    float rstd = rsqrtf(var + 1e-5f);
#pragma unroll
    for (int f = 0; f < 4; f++) {
        int n = lane + f * 32;
        float o = gelu((y[f] - mean) * rstd * g[n] + be[n]);
        h[(size_t)wid * HIDDEN + n]  = o;
        hb[(size_t)wid * HIDDEN + n] = f2h(o);
    }
}

// Fused edge kernel: msg = (GELU(LN([h[row],ea] @ W1 + b1))) @ W2 + b2; atomic scatter to agg[col]
// LDS: packed W1 frags (64KB) | packed W2 frags (64KB) | per-wave H1 staging (8*8KB)
__global__ void __launch_bounds__(256) k_edge(
    const int* __restrict__ ei, const float* __restrict__ eattr,
    const uint16_t* __restrict__ hb,
    const float* __restrict__ W1, const float* __restrict__ b1,
    const float* __restrict__ g1, const float* __restrict__ be1,
    const float* __restrict__ W2, const float* __restrict__ b2,
    float* __restrict__ agg) {
    extern __shared__ uint32_t lds[];
    uint32_t* w1p = lds;              // 64 frags * 256 dwords
    uint32_t* w2p = lds + 16384;      // 64 frags * 256 dwords
    uint16_t* h1s = (uint16_t*)(lds + 32768);  // 8 waves * 16*256 f16

    // pack B fragments (f32 -> packed f16 pairs); B layout: lane=N, VGPR v holds K pair
    for (int idx = threadIdx.x; idx < 16384; idx += 256) {
        int frag = idx >> 8, rem = idx & 255, ln = rem >> 3, v = rem & 7;
        int kc = frag >> 4, nt = frag & 15;
        int n = nt * 16 + (ln & 15);
        int k = kc * 32 + ((ln >> 4) << 3) + ((v & 3) << 1) + ((v >> 2) << 4);
        w1p[idx] = pk2h(W1[k * 256 + n], W1[(k + 1) * 256 + n]);
    }
    for (int idx = threadIdx.x; idx < 16384; idx += 256) {
        int frag = idx >> 8, rem = idx & 255, ln = rem >> 3, v = rem & 7;
        int kc = frag >> 3, nt = frag & 7;
        int n = nt * 16 + (ln & 15);
        int k = kc * 32 + ((ln >> 4) << 3) + ((v & 3) << 1) + ((v >> 2) << 4);
        w2p[idx] = pk2h(W2[k * 128 + n], W2[(k + 1) * 128 + n]);
    }
    __syncthreads();

    const int lane = threadIdx.x & 31;
    const int wv   = threadIdx.x >> 5;
    const int l15  = lane & 15;
    const int hi   = lane >> 4;
    uint16_t* hs = h1s + wv * 4096;

    float bias1[16], gam1[16], bet1[16], w1e0[16], w1e1[16];
#pragma unroll
    for (int nt = 0; nt < 16; nt++) {
        int n = nt * 16 + l15;
        bias1[nt] = b1[n]; gam1[nt] = g1[n]; bet1[nt] = be1[n];
        w1e0[nt] = W1[128 * 256 + n]; w1e1[nt] = W1[129 * 256 + n];
    }
    float bias2[8];
#pragma unroll
    for (int nt = 0; nt < 8; nt++) bias2[nt] = b2[nt * 16 + l15];

    const int gw = blockIdx.x * 8 + wv;
    const int tw = gridDim.x * 8;

    for (int t = gw; t < N_EDGES / 16; t += tw) {
        const int e    = t * 16 + l15;
        const int r    = ei[e];
        const int cdst = ei[N_EDGES + e];
        const float ea0 = eattr[2 * e], ea1 = eattr[2 * e + 1];

        float e0v[8], e1v[8]; int cj[8];
#pragma unroll
        for (int j = 0; j < 8; j++) {
            int m = j + hi * 8;
            e0v[j] = __shfl(ea0, m);
            e1v[j] = __shfl(ea1, m);
            cj[j]  = __shfl(cdst, m);
        }

        // A fragments: gather f16 node features (L2-resident), 2x b128 per 32-chunk
        uint32_t a1[4][8];
        const uint16_t* hr = hb + (size_t)r * HIDDEN;
#pragma unroll
        for (int kc = 0; kc < 4; kc++) ldA(hr, kc, hi, a1[kc]);

        // C init: bias + rank-2 edge-attr contribution (rows 128,129 of W1)
        floatx8 c1[16];
#pragma unroll
        for (int nt = 0; nt < 16; nt++)
#pragma unroll
            for (int j = 0; j < 8; j++)
                c1[nt][j] = bias1[nt] + e0v[j] * w1e0[nt] + e1v[j] * w1e1[nt];

        // GEMM1: flattened frag loop, double-buffered B fragments (1-frag lookahead)
        {
            uint32_t bbuf[2][8];
            ldB(w1p + lane * 8, bbuf[0]);
#pragma unroll
            for (int f = 0; f < 64; f++) {
                if (f < 63) ldB(w1p + ((f + 1) << 8) + lane * 8, bbuf[(f + 1) & 1]);
                c1[f & 15] = wmma16(a1[f >> 4], bbuf[f & 1], c1[f & 15]);
            }
        }

        // LN over 256 + GELU, stage to LDS as f16 A-tile for GEMM2
#pragma unroll
        for (int j = 0; j < 8; j++) {
            float s = 0.f, s2 = 0.f;
#pragma unroll
            for (int nt = 0; nt < 16; nt++) { float vv = c1[nt][j]; s += vv; s2 += vv * vv; }
#pragma unroll
            for (int m = 1; m < 16; m <<= 1) { s += __shfl_xor(s, m); s2 += __shfl_xor(s2, m); }
            float mean = s * (1.f / 256.f);
            float var  = s2 * (1.f / 256.f) - mean * mean;
            float rstd = rsqrtf(var + 1e-5f);
            int mrow = j + hi * 8;
#pragma unroll
            for (int nt = 0; nt < 16; nt++) {
                float o = gelu((c1[nt][j] - mean) * rstd * gam1[nt] + bet1[nt]);
                hs[mrow * 256 + nt * 16 + l15] = f2h(o);
            }
        }
        asm volatile("s_wait_dscnt 0" ::: "memory");

        // GEMM2: [16,256] @ W2[256,128]; double-buffered A and B fragments
        floatx8 c2[8];
#pragma unroll
        for (int nt = 0; nt < 8; nt++)
#pragma unroll
            for (int j = 0; j < 8; j++) c2[nt][j] = bias2[nt];
        {
            uint32_t abuf[2][8], bbuf[2][8];
            ldA(hs + l15 * 256, 0, hi, abuf[0]);
            ldB(w2p + lane * 8, bbuf[0]);
#pragma unroll
            for (int f = 0; f < 64; f++) {
                int kc = f >> 3;
                if (f < 63) {
                    ldB(w2p + ((f + 1) << 8) + lane * 8, bbuf[(f + 1) & 1]);
                    if ((f & 7) == 7)
                        ldA(hs + l15 * 256, kc + 1, hi, abuf[(kc + 1) & 1]);
                }
                c2[f & 7] = wmma16(abuf[kc & 1], bbuf[f & 1], c2[f & 7]);
            }
        }

        // scatter-add into L2-resident agg
#pragma unroll
        for (int nt = 0; nt < 8; nt++)
#pragma unroll
            for (int j = 0; j < 8; j++)
                atomicAdd(agg + (size_t)cj[j] * HIDDEN + nt * 16 + l15, c2[nt][j]);
    }
}

// Node update: h = GELU(LN([h,agg] @ updW + b)) + h ; also per-layer mean-pool sums
__global__ void __launch_bounds__(256) k_node(
    const float* __restrict__ uW, const float* __restrict__ ub,
    const float* __restrict__ ug, const float* __restrict__ ube,
    const int* __restrict__ batch, const float* __restrict__ agg,
    float* __restrict__ h, uint16_t* __restrict__ hb, float* __restrict__ pool) {
    extern __shared__ uint32_t lds[];
    for (int idx = threadIdx.x; idx < 16384; idx += 256) {
        int frag = idx >> 8, rem = idx & 255, ln = rem >> 3, v = rem & 7;
        int kc = frag >> 3, nt = frag & 7;
        int n = nt * 16 + (ln & 15);
        int k = kc * 32 + ((ln >> 4) << 3) + ((v & 3) << 1) + ((v >> 2) << 4);
        lds[idx] = pk2h(uW[k * 128 + n], uW[(k + 1) * 128 + n]);
    }
    __syncthreads();
    const int lane = threadIdx.x & 31, wv = threadIdx.x >> 5, l15 = lane & 15, hi = lane >> 4;
    float bz[8], gz[8], bb[8];
#pragma unroll
    for (int nt = 0; nt < 8; nt++) {
        int n = nt * 16 + l15;
        bz[nt] = ub[n]; gz[nt] = ug[n]; bb[nt] = ube[n];
    }
    const int gw = blockIdx.x * 8 + wv, tw = gridDim.x * 8;
    for (int t = gw; t < N_NODES / 16; t += tw) {
        int node = t * 16 + l15;
        int bt = batch[node];
        int bj[8];
#pragma unroll
        for (int j = 0; j < 8; j++) bj[j] = __shfl(bt, j + hi * 8);

        uint32_t a[8][8];
        const uint16_t* hr = hb + (size_t)node * HIDDEN;
#pragma unroll
        for (int kc = 0; kc < 4; kc++) ldA(hr, kc, hi, a[kc]);
        // agg half: f32 -> f16 pack, 4x b128 float loads per 32-chunk
        const float* ar = agg + (size_t)node * HIDDEN;
#pragma unroll
        for (int kc = 4; kc < 8; kc++) {
            int kb = (kc - 4) * 32 + hi * 8;
            float4 f0 = *(const float4*)(ar + kb);
            float4 f1 = *(const float4*)(ar + kb + 4);
            float4 f2 = *(const float4*)(ar + kb + 16);
            float4 f3 = *(const float4*)(ar + kb + 20);
            a[kc][0] = pk2h(f0.x, f0.y); a[kc][1] = pk2h(f0.z, f0.w);
            a[kc][2] = pk2h(f1.x, f1.y); a[kc][3] = pk2h(f1.z, f1.w);
            a[kc][4] = pk2h(f2.x, f2.y); a[kc][5] = pk2h(f2.z, f2.w);
            a[kc][6] = pk2h(f3.x, f3.y); a[kc][7] = pk2h(f3.z, f3.w);
        }

        floatx8 c[8];
#pragma unroll
        for (int nt = 0; nt < 8; nt++)
#pragma unroll
            for (int j = 0; j < 8; j++) c[nt][j] = bz[nt];
        // flattened frag loop, double-buffered B fragments
        {
            uint32_t bbuf[2][8];
            ldB(lds + lane * 8, bbuf[0]);
#pragma unroll
            for (int f = 0; f < 64; f++) {
                if (f < 63) ldB(lds + ((f + 1) << 8) + lane * 8, bbuf[(f + 1) & 1]);
                c[f & 7] = wmma16(a[f >> 3], bbuf[f & 1], c[f & 7]);
            }
        }

#pragma unroll
        for (int j = 0; j < 8; j++) {
            float s = 0.f, s2 = 0.f;
#pragma unroll
            for (int nt = 0; nt < 8; nt++) { float vv = c[nt][j]; s += vv; s2 += vv * vv; }
#pragma unroll
            for (int m = 1; m < 16; m <<= 1) { s += __shfl_xor(s, m); s2 += __shfl_xor(s2, m); }
            float mean = s * (1.f / 128.f);
            float var  = s2 * (1.f / 128.f) - mean * mean;
            float rstd = rsqrtf(var + 1e-5f);
            size_t mnode = (size_t)(t * 16 + j + hi * 8);
#pragma unroll
            for (int nt = 0; nt < 8; nt++) {
                int n = nt * 16 + l15;
                float o = gelu((c[nt][j] - mean) * rstd * gz[nt] + bb[nt]) + h[mnode * HIDDEN + n];
                h[mnode * HIDDEN + n]  = o;
                hb[mnode * HIDDEN + n] = f2h(o);
                atomicAdd(pool + bj[j] * HIDDEN + n, o);
            }
        }
    }
}

// g = GELU(LN(concat(layer_means) @ combine_W + b))
__global__ void k_combine(const float* __restrict__ W, const float* __restrict__ b,
                          const float* __restrict__ be, const float* __restrict__ g,
                          const float* __restrict__ pool, const float* __restrict__ cnt,
                          float* __restrict__ out) {
    __shared__ float z[NUM_GRAPHS * HIDDEN];
    __shared__ float mu[NUM_GRAPHS], rs[NUM_GRAPHS];
    int t = threadIdx.x;
    for (int idx = t; idx < NUM_GRAPHS * HIDDEN; idx += 256) {
        int gi = idx >> 7, n = idx & 127;
        float inv = 1.0f / fmaxf(cnt[gi], 1.0f);
        float acc = b[n];
        for (int l = 0; l < 4; l++)
            for (int k = 0; k < 128; k++)
                acc += pool[l * 8192 + gi * 128 + k] * inv * W[(l * 128 + k) * 128 + n];
        z[idx] = acc;
    }
    __syncthreads();
    if (t < NUM_GRAPHS) {
        float s = 0.f, s2 = 0.f;
        for (int k = 0; k < 128; k++) { float v = z[t * 128 + k]; s += v; s2 += v * v; }
        float mean = s * (1.f / 128.f);
        float var  = s2 * (1.f / 128.f) - mean * mean;
        mu[t] = mean; rs[t] = rsqrtf(var + 1e-5f);
    }
    __syncthreads();
    for (int idx = t; idx < NUM_GRAPHS * HIDDEN; idx += 256) {
        int gi = idx >> 7, n = idx & 127;
        out[idx] = gelu((z[idx] - mu[gi]) * rs[gi] * g[n] + be[n]);
    }
}

extern "C" void kernel_launch(void* const* d_in, const int* in_sizes, int n_in,
                              void* d_out, int out_size, void* d_ws, size_t ws_size,
                              hipStream_t stream) {
    (void)in_sizes; (void)n_in; (void)out_size; (void)ws_size;
    const float* x      = (const float*)d_in[0];
    const int*   ei     = (const int*)d_in[1];
    const float* eattr  = (const float*)d_in[2];
    const int*   batch  = (const int*)d_in[3];
    // params flattened alphabetically (JAX pytree dict order)
    const float* cmb_W  = (const float*)d_in[4];
    const float* cmb_b  = (const float*)d_in[5];
    const float* cmb_be = (const float*)d_in[6];
    const float* cmb_g  = (const float*)d_in[7];
    const float* emb_W  = (const float*)d_in[8];
    const float* emb_b  = (const float*)d_in[9];
    const float* emb_be = (const float*)d_in[10];
    const float* emb_g  = (const float*)d_in[11];

    char* ws = (char*)d_ws;
    uint16_t* hb  = (uint16_t*)ws;                      // 12,800,000 B
    float* agg    = (float*)(ws + 12800000);            // 25,600,000 B
    float* pool   = (float*)(ws + 38400000);            // 4*64*128 floats
    float* cnt    = pool + 4 * NUM_GRAPHS * HIDDEN;     // 64 floats (contiguous after pool)
    float* g_out  = (float*)d_out;
    float* h      = g_out + NUM_GRAPHS * HIDDEN;        // h lives in its output slot

    k_zero<<<(4 * NUM_GRAPHS * HIDDEN + NUM_GRAPHS + 255) / 256, 256, 0, stream>>>(
        pool, 4 * NUM_GRAPHS * HIDDEN + NUM_GRAPHS);
    k_count<<<(N_NODES + 255) / 256, 256, 0, stream>>>(batch, cnt);
    k_embed<<<(N_NODES * 32 + 255) / 256, 256, 0, stream>>>(x, emb_W, emb_b, emb_g, emb_be, h, hb);

    for (int l = 0; l < 4; l++) {
        const float* mW1 = (const float*)d_in[12 + 10 * l + 0];
        const float* mW2 = (const float*)d_in[12 + 10 * l + 1];
        const float* mb1 = (const float*)d_in[12 + 10 * l + 2];
        const float* mb2 = (const float*)d_in[12 + 10 * l + 3];
        const float* mbe = (const float*)d_in[12 + 10 * l + 4];
        const float* mg  = (const float*)d_in[12 + 10 * l + 5];
        const float* uW  = (const float*)d_in[12 + 10 * l + 6];
        const float* ubb = (const float*)d_in[12 + 10 * l + 7];
        const float* ube = (const float*)d_in[12 + 10 * l + 8];
        const float* ugg = (const float*)d_in[12 + 10 * l + 9];

        k_zero<<<(N_NODES * HIDDEN + 255) / 256, 256, 0, stream>>>(agg, N_NODES * HIDDEN);
        k_edge<<<800, 256, 196608, stream>>>(ei, eattr, hb, mW1, mb1, mg, mbe, mW2, mb2, agg);
        k_node<<<200, 256, 65536, stream>>>(uW, ubb, ugg, ube, batch, agg, h, hb,
                                            pool + l * NUM_GRAPHS * HIDDEN);
    }
    k_combine<<<1, 256, 0, stream>>>(cmb_W, cmb_b, cmb_be, cmb_g, pool, cnt, g_out);
}
